// DGMNet_76484777607677
// MI455X (gfx1250) — matrix operand order repaired and checked by reading, not compiled, not run on hardware
//
#include <hip/hip_runtime.h>
#include <cstdint>

#define DEV __device__ __forceinline__

typedef __attribute__((ext_vector_type(16))) __bf16 v16bf;
typedef __attribute__((ext_vector_type(8)))  float  v8f;

// ---- problem constants ----
constexpr int BATCH   = 65536;
constexpr int DIN     = 101;    // true input width
constexpr int KX      = 128;    // padded input width (K for X@U)
constexpr int KXP     = KX / 2; // packed u32 per X row = 64
constexpr int KXB     = KX / 32;  // 4 K-blocks for X-side GEMMs
constexpr int NN      = 512;    // hidden width
constexpr int NNP     = NN / 2; // 256 packed u32 per S row
constexpr int NNB     = NN / 32;  // 16 K-blocks for S-side GEMMs
constexpr int NTILES  = NN / 16;  // 32 n-tiles
constexpr int NLAYERS = 3;
constexpr int MT      = 64;     // rows per workgroup

// Swizzled weight layout: frag(ntile, kb) = 32 lanes x 8 u32 (bf16 pairs),
// lane-contiguous so each lane issues two global_load_b128.
// index = ((ntile*K32 + kb)*32 + lane)*8 + v
// value  = bf16pair( W[2*kp, n], W[2*kp+1, n] ),  kp = kb*16 + v + 8*(lane>>4),
//          n = ntile*16 + (lane&15)

// ---- workspace layout (units: uint32) ----
constexpr size_t OFF_XP = 0;
constexpr size_t SZ_XP  = (size_t)BATCH * KXP;        // 4,194,304 u32
constexpr size_t OFF_W0 = OFF_XP + SZ_XP;
constexpr size_t SZ_W0  = (size_t)NTILES * KXB * 256; // 32,768 u32
constexpr size_t OFF_U  = OFF_W0 + SZ_W0;
constexpr size_t SZ_U1  = (size_t)NTILES * KXB * 256; // per (gate,layer)
constexpr size_t OFF_W  = OFF_U + 12 * SZ_U1;
constexpr size_t SZ_W1  = (size_t)NTILES * NNB * 256; // 131,072 per (gate,layer)

// LDS row strides (u32), padded to spread rows across banks
constexpr int XLD = KXP + 4;   // 68
constexpr int SLD = NNP + 4;   // 260

DEV uint16_t f2bf(float f) {
  uint32_t u = __float_as_uint(f);
  u += 0x7FFFu + ((u >> 16) & 1u);          // round-to-nearest-even
  return (uint16_t)(u >> 16);
}
DEV float bf2f(uint16_t h) { return __uint_as_float(((uint32_t)h) << 16); }
DEV uint32_t pack2(float lo, float hi) {
  return (uint32_t)f2bf(lo) | ((uint32_t)f2bf(hi) << 16);
}

// Branch-free tanh: v_exp_f32 + v_rcp_f32, saturates correctly at +-inf,
// exact at 0, NaN-propagating. No exec-divergent slow path.
DEV float tanh_fast(float x) {
  float e = __expf(2.0f * x);
  return 1.0f - __fdividef(2.0f, e + 1.0f);
}

union ABFrag { uint32_t u[8]; v16bf v; };

// acc[mt] += A(mt-tile from LDS) * B(ntile from swizzled global), K = 32*nkb
DEV void gemm_acc(v8f acc[4], const uint32_t* __restrict__ Bsw, int ntile,
                  const uint32_t* Alds, int lds_stride, int nkb,
                  int laneM, int half, int lane) {
  const uint32_t* bbase = Bsw + ((size_t)ntile * nkb) * 256 + lane * 8;
#pragma unroll 4
  for (int kb = 0; kb < nkb; ++kb) {
    const int kpB = kb * 16;
    ABFrag b;
    *(uint4*)(&b.u[0]) = *(const uint4*)(bbase);       // 32B contiguous per lane
    *(uint4*)(&b.u[4]) = *(const uint4*)(bbase + 4);
    bbase += 256;
#pragma unroll
    for (int mt = 0; mt < 4; ++mt) {
      const uint32_t* ab = Alds + (size_t)(mt * 16 + laneM) * lds_stride + kpB + 8 * half;
      ABFrag a;
      *(uint4*)(&a.u[0]) = *(const uint4*)(ab);       // K pairs 8h+0..3
      *(uint4*)(&a.u[4]) = *(const uint4*)(ab + 4);   // K pairs 8h+4..7
      acc[mt] = __builtin_amdgcn_wmma_f32_16x16x32_bf16(
          false, a.v, false, b.v, (short)0, acc[mt], false, false);
    }
  }
}

// ---- packing pre-pass kernels ----
__global__ void pack_x_kernel(const float* __restrict__ X, uint32_t* __restrict__ dst) {
  size_t idx = (size_t)blockIdx.x * blockDim.x + threadIdx.x;
  if (idx >= SZ_XP) return;
  int row = (int)(idx >> 6);
  int kp  = (int)(idx & 63);
  int k0 = kp * 2, k1 = k0 + 1;
  float lo = (k0 < DIN) ? X[(size_t)row * DIN + k0] : 0.f;
  float hi = (k1 < DIN) ? X[(size_t)row * DIN + k1] : 0.f;
  dst[idx] = pack2(lo, hi);
}

// src: fp32 [K][512] row-major (valid K = Ka, padded to 32*K32 with zeros)
// dst: swizzled fragment layout described above
__global__ void pack_w_swz_kernel(const float* __restrict__ src, uint32_t* __restrict__ dst,
                                  int Ka, int K32) {
  size_t total = (size_t)NTILES * K32 * 256;
  size_t idx = (size_t)blockIdx.x * blockDim.x + threadIdx.x;
  if (idx >= total) return;
  int v     = (int)(idx & 7);
  int lane  = (int)((idx >> 3) & 31);
  int kb    = (int)((idx >> 8) % K32);
  int ntile = (int)(idx / ((size_t)K32 * 256));
  int laneM = lane & 15;
  int half  = lane >> 4;
  int kp = kb * 16 + v + 8 * half;
  int n  = ntile * 16 + laneM;
  int k0 = kp * 2, k1 = k0 + 1;
  float lo = (k0 < Ka) ? src[(size_t)k0 * NN + n] : 0.f;
  float hi = (k1 < Ka) ? src[(size_t)k1 * NN + n] : 0.f;
  dst[idx] = pack2(lo, hi);
}

// ---- fused DGM forward ----
__global__ __launch_bounds__(256) void dgm_main(
    const uint32_t* __restrict__ ws,
    const float* __restrict__ b0,
    const float* __restrict__ bz, const float* __restrict__ bg,
    const float* __restrict__ br, const float* __restrict__ bh,
    const float* __restrict__ Wf, const float* __restrict__ bfin,
    float* __restrict__ out) {
  __shared__ uint32_t Xs[MT * XLD];      // 17 KB
  __shared__ uint32_t Sbuf0[MT * SLD];   // 65 KB
  __shared__ uint32_t Sbuf1[MT * SLD];   // 65 KB

  const int tid   = threadIdx.x;
  const int wave  = tid >> 5;
  const int lane  = tid & 31;
  const int laneM = lane & 15;
  const int half  = lane >> 4;
  const int row0  = blockIdx.x * MT;

  // stage packed X tile
  for (int idx = tid; idx < MT * KXP; idx += 256) {
    int r = idx >> 6, c = idx & 63;
    Xs[r * XLD + c] = ws[OFF_XP + (size_t)(row0 + r) * KXP + c];
  }
  __syncthreads();

  uint32_t* cur = Sbuf0;
  uint32_t* nxt = Sbuf1;

  // initial dense: S = tanh(X @ W0 + b0)
  {
    uint16_t* s0h = (uint16_t*)cur;
    for (int nt = 0; nt < 4; ++nt) {
      const int ntile = wave * 4 + nt;
      const int n     = ntile * 16 + laneM;
      v8f acc[4];
#pragma unroll
      for (int mt = 0; mt < 4; ++mt) acc[mt] = (v8f)(0.0f);
      gemm_acc(acc, ws + OFF_W0, ntile, Xs, XLD, KXB, laneM, half, lane);
      float bias = b0[n];
#pragma unroll
      for (int mt = 0; mt < 4; ++mt)
#pragma unroll
        for (int v = 0; v < 8; ++v) {
          int m = mt * 16 + v + 8 * half;
          s0h[m * (2 * SLD) + n] = f2bf(tanh_fast(acc[mt][v] + bias));
        }
    }
  }
  __syncthreads();

  for (int L = 0; L < NLAYERS; ++L) {
    const uint32_t* Uzp = ws + OFF_U + (size_t)(0 * 3 + L) * SZ_U1;
    const uint32_t* Ugp = ws + OFF_U + (size_t)(1 * 3 + L) * SZ_U1;
    const uint32_t* Urp = ws + OFF_U + (size_t)(2 * 3 + L) * SZ_U1;
    const uint32_t* Uhp = ws + OFF_U + (size_t)(3 * 3 + L) * SZ_U1;
    const uint32_t* Wzp = ws + OFF_W + (size_t)(0 * 3 + L) * SZ_W1;
    const uint32_t* Wgp = ws + OFF_W + (size_t)(1 * 3 + L) * SZ_W1;
    const uint32_t* Wrp = ws + OFF_W + (size_t)(2 * 3 + L) * SZ_W1;
    const uint32_t* Whp = ws + OFF_W + (size_t)(3 * 3 + L) * SZ_W1;
    const float* bzl = bz + L * NN;
    const float* bgl = bg + L * NN;
    const float* brl = br + L * NN;
    const float* bhl = bh + L * NN;
    uint16_t* curh = (uint16_t*)cur;
    uint16_t* nxth = (uint16_t*)nxt;

    // pass A: R = tanh(X@Ur + S@Wr + br); stage SR = S*R into nxt
    for (int nt = 0; nt < 4; ++nt) {
      const int ntile = wave * 4 + nt;
      const int n     = ntile * 16 + laneM;
      v8f acc[4];
#pragma unroll
      for (int mt = 0; mt < 4; ++mt) acc[mt] = (v8f)(0.0f);
      gemm_acc(acc, Urp, ntile, Xs, XLD, KXB, laneM, half, lane);
      gemm_acc(acc, Wrp, ntile, cur, SLD, NNB, laneM, half, lane);
      float bias = brl[n];
#pragma unroll
      for (int mt = 0; mt < 4; ++mt)
#pragma unroll
        for (int v = 0; v < 8; ++v) {
          int m = mt * 16 + v + 8 * half;
          float r  = tanh_fast(acc[mt][v] + bias);
          float so = bf2f(curh[m * (2 * SLD) + n]);
          nxth[m * (2 * SLD) + n] = f2bf(so * r);
        }
    }
    __syncthreads();

    // pass B1: H = tanh(X@Uh + SR@Wh + bh), held in registers
    float H[4][4][8];
    for (int nt = 0; nt < 4; ++nt) {
      const int ntile = wave * 4 + nt;
      const int n     = ntile * 16 + laneM;
      v8f acc[4];
#pragma unroll
      for (int mt = 0; mt < 4; ++mt) acc[mt] = (v8f)(0.0f);
      gemm_acc(acc, Uhp, ntile, Xs, XLD, KXB, laneM, half, lane);
      gemm_acc(acc, Whp, ntile, nxt, SLD, NNB, laneM, half, lane);
      float bias = bhl[n];
#pragma unroll
      for (int mt = 0; mt < 4; ++mt)
#pragma unroll
        for (int v = 0; v < 8; ++v)
          H[nt][mt][v] = tanh_fast(acc[mt][v] + bias);
    }
    __syncthreads();  // all SR reads done; nxt may now be overwritten

    // pass B2: Z, G; S_new = (1-G)*H + Z*S  -> nxt
    for (int nt = 0; nt < 4; ++nt) {
      const int ntile = wave * 4 + nt;
      const int n     = ntile * 16 + laneM;
      float P[4][8], OM[4][8];
      {
        v8f acc[4];
#pragma unroll
        for (int mt = 0; mt < 4; ++mt) acc[mt] = (v8f)(0.0f);
        gemm_acc(acc, Uzp, ntile, Xs, XLD, KXB, laneM, half, lane);
        gemm_acc(acc, Wzp, ntile, cur, SLD, NNB, laneM, half, lane);
        float bias = bzl[n];
#pragma unroll
        for (int mt = 0; mt < 4; ++mt)
#pragma unroll
          for (int v = 0; v < 8; ++v) {
            int m = mt * 16 + v + 8 * half;
            float z = tanh_fast(acc[mt][v] + bias);
            P[mt][v] = z * bf2f(curh[m * (2 * SLD) + n]);
          }
      }
      {
        v8f acc[4];
#pragma unroll
        for (int mt = 0; mt < 4; ++mt) acc[mt] = (v8f)(0.0f);
        gemm_acc(acc, Ugp, ntile, Xs, XLD, KXB, laneM, half, lane);
        gemm_acc(acc, Wgp, ntile, cur, SLD, NNB, laneM, half, lane);
        float bias = bgl[n];
#pragma unroll
        for (int mt = 0; mt < 4; ++mt)
#pragma unroll
          for (int v = 0; v < 8; ++v)
            OM[mt][v] = 1.0f - tanh_fast(acc[mt][v] + bias);
      }
#pragma unroll
      for (int mt = 0; mt < 4; ++mt)
#pragma unroll
        for (int v = 0; v < 8; ++v) {
          int m = mt * 16 + v + 8 * half;
          nxth[m * (2 * SLD) + n] = f2bf(OM[mt][v] * H[nt][mt][v] + P[mt][v]);
        }
    }
    __syncthreads();
    uint32_t* t = cur; cur = nxt; nxt = t;
  }

  // final: out = S @ Wf + bf
  if (tid < MT) {
    const uint16_t* sh = (const uint16_t*)cur;
    float s = bfin[0];
    for (int k = 0; k < NN; ++k)
      s += bf2f(sh[tid * (2 * SLD) + k]) * Wf[k];
    out[row0 + tid] = s;
  }
}

extern "C" void kernel_launch(void* const* d_in, const int* in_sizes, int n_in,
                              void* d_out, int out_size, void* d_ws, size_t ws_size,
                              hipStream_t stream) {
  (void)in_sizes; (void)n_in; (void)out_size; (void)ws_size;
  const float* X  = (const float*)d_in[0];
  const float* W0 = (const float*)d_in[1];
  const float* b0 = (const float*)d_in[2];
  const float* Uz = (const float*)d_in[3];
  const float* Wz = (const float*)d_in[4];
  const float* bz = (const float*)d_in[5];
  const float* Ug = (const float*)d_in[6];
  const float* Wg = (const float*)d_in[7];
  const float* bg = (const float*)d_in[8];
  const float* Ur = (const float*)d_in[9];
  const float* Wr = (const float*)d_in[10];
  const float* br = (const float*)d_in[11];
  const float* Uh = (const float*)d_in[12];
  const float* Wh = (const float*)d_in[13];
  const float* bh = (const float*)d_in[14];
  const float* Wf = (const float*)d_in[15];
  const float* bf = (const float*)d_in[16];
  float* out   = (float*)d_out;
  uint32_t* ws = (uint32_t*)d_ws;

  pack_x_kernel<<<(int)((SZ_XP + 255) / 256), 256, 0, stream>>>(X, ws + OFF_XP);
  pack_w_swz_kernel<<<(int)((SZ_W0 + 255) / 256), 256, 0, stream>>>(
      W0, ws + OFF_W0, DIN, KXB);

  const float* Uarr[4] = {Uz, Ug, Ur, Uh};
  const float* Warr[4] = {Wz, Wg, Wr, Wh};
  for (int g = 0; g < 4; ++g)
    for (int L = 0; L < NLAYERS; ++L) {
      pack_w_swz_kernel<<<(int)((SZ_U1 + 255) / 256), 256, 0, stream>>>(
          Uarr[g] + (size_t)L * DIN * NN, ws + OFF_U + (size_t)(g * 3 + L) * SZ_U1,
          DIN, KXB);
      pack_w_swz_kernel<<<(int)((SZ_W1 + 255) / 256), 256, 0, stream>>>(
          Warr[g] + (size_t)L * NN * NN, ws + OFF_W + (size_t)(g * 3 + L) * SZ_W1,
          NN, NNB);
    }

  dgm_main<<<BATCH / MT, 256, 0, stream>>>(ws, b0, bz, bg, br, bh, Wf, bf, out);
}
